// MagNet_25125558682188
// MI455X (gfx1250) — compile-verified, matrix operand order checked by reading, not run on hardware
//
#include <hip/hip_runtime.h>

// ===========================================================================
// MagNet forward for MI455X (gfx1250, wave32, WMMA).
// All 1x1 convs and the attn@v einsum run through a f16-input / f32-accum
// WMMA GEMM (v_wmma_f32_16x16x32_f16) with vectorized, guard-free staging and
// a double-buffered B tile in LDS. hq@k^T uses a split-K WMMA kernel with f32
// global atomics. l2norm is folded into the attention epilogue as per-row
// norms. Bias/GELU/residual fused into GEMM epilogues.
// ===========================================================================

typedef __attribute__((ext_vector_type(16))) _Float16 v16h;
typedef __attribute__((ext_vector_type(8)))  float    v8f;
typedef __attribute__((ext_vector_type(4)))  _Float16 h4;

#define NB 4  // batch

__device__ __forceinline__ float gelu_f(float x) {
    return 0.5f * x * (1.0f + erff(x * 0.70710678118654752440f));
}

// ---------------------------------------------------------------------------
// Generic WMMA GEMM:  Y[m,n] = act( sum_k W[m,k] * X[k,n] + bias[m] ) + R[m,n]
// Per-z operand base:  P += (z/zdiv)*b1 + (z%zdiv)*b2
// Block: 256 thr = 8 waves, tile M=16, N=128, K-step 32.
// REQUIRES (guaranteed by all call sites): M % 16 == 0 (grid.y == M/16),
// N % 128 == 0 (grid.x == N/128), K % 4 == 0, K <= 384.
// ---------------------------------------------------------------------------
__global__ __launch_bounds__(256)
void k_gemm16(const float* __restrict__ X, long long xb1, long long xb2,
              const float* __restrict__ Wm, long long wb1, long long wb2,
              const float* __restrict__ bias,
              const float* __restrict__ R, long long rb1, long long rb2,
              float* __restrict__ Y, long long yb1, long long yb2,
              int M, int K, int N, int zdiv, int act)
{
    const int z = blockIdx.z;
    const long long zq = z / zdiv, zr = z % zdiv;
    X  += zq * xb1 + zr * xb2;
    Wm += zq * wb1 + zr * wb2;
    if (R) R += zq * rb1 + zr * rb2;
    Y  += zq * yb1 + zr * yb2;

    const int Nb = blockIdx.x * 128;
    const int Mb = blockIdx.y * 16;
    const int Kpad = (K + 31) & ~31;
    const int nt   = Kpad >> 5;          // number of 32-wide K tiles
    const int lda  = Kpad + 8;           // A row stride (halves): 16B-aligned, bank-spread

    __shared__ _Float16 As[16 * (384 + 8)];
    __shared__ _Float16 Bs[2][128][40];  // [n][k], 80B rows: aligned + bank-spread

    const int tid  = threadIdx.x;
    const int wv   = tid >> 5;           // wave -> N sub-tile
    const int lane = tid & 31;
    const int l16  = lane & 15;
    const int hi   = lane >> 4;
    const int kb   = hi << 3;

    // ---- stage the 16 x Kpad A (weight) panel once, f32 -> f16, float4 wide ----
    {
        const int q4 = Kpad >> 2;        // quads per row
        for (int q = tid; q < 16 * q4; q += 256) {
            const int mr = q / q4, kq = (q - mr * q4) << 2;
            h4 c = {};
            if (kq < K) {                // K % 4 == 0 -> whole quad valid
                const float4 v = *(const float4*)&Wm[(long long)(Mb + mr) * K + kq];
                c[0] = (_Float16)v.x; c[1] = (_Float16)v.y;
                c[2] = (_Float16)v.z; c[3] = (_Float16)v.w;
            }
            *(h4*)&As[mr * lda + kq] = c;
        }
    }

    // ---- B tile staging: registers <-> LDS ping-pong ----
    float4 rb[4];
    auto ldB = [&](int t) {
        const int k0 = t << 5;
        if (k0 + 32 <= K) {              // full tile: no guards at all
#pragma unroll
            for (int i = 0; i < 4; ++i) {
                const int idx = tid + (i << 8);
                const int kc = idx >> 5, nc = (idx & 31) << 2;
                rb[i] = *(const float4*)&X[(long long)(k0 + kc) * N + Nb + nc];
            }
        } else {                         // tail tile (only K % 32 == 16 case)
#pragma unroll
            for (int i = 0; i < 4; ++i) {
                const int idx = tid + (i << 8);
                const int kc = idx >> 5, nc = (idx & 31) << 2;
                if (k0 + kc < K)
                    rb[i] = *(const float4*)&X[(long long)(k0 + kc) * N + Nb + nc];
                else { rb[i].x = rb[i].y = rb[i].z = rb[i].w = 0.0f; }
            }
        }
    };
    auto stB = [&](int t) {
        const int buf = t & 1;
#pragma unroll
        for (int i = 0; i < 4; ++i) {
            const int idx = tid + (i << 8);
            const int kc = idx >> 5, nc = (idx & 31) << 2;
            Bs[buf][nc + 0][kc] = (_Float16)rb[i].x;
            Bs[buf][nc + 1][kc] = (_Float16)rb[i].y;
            Bs[buf][nc + 2][kc] = (_Float16)rb[i].z;
            Bs[buf][nc + 3][kc] = (_Float16)rb[i].w;
        }
    };

    v8f acc = {};
    ldB(0); stB(0);
    __syncthreads();
    for (int t = 0; t < nt; ++t) {
        if (t + 1 < nt) {
            ldB(t + 1);                  // overlap next tile's HBM stream with WMMA
            if (t + 2 < nt)              // speculative prefetch (global_prefetch_b8)
                __builtin_prefetch(&X[(long long)((t + 2) << 5) * N + Nb + (tid & 127)], 0, 1);
        }
        const int k0 = t << 5, buf = t & 1;
        v16h a, b;
#pragma unroll
        for (int e = 0; e < 8; ++e) {
            a[e]     = As[l16 * lda + k0 + kb + e];
            a[e + 8] = As[l16 * lda + k0 + 16 + kb + e];
            b[e]     = Bs[buf][wv * 16 + l16][kb + e];
            b[e + 8] = Bs[buf][wv * 16 + l16][16 + kb + e];
        }
        acc = __builtin_amdgcn_wmma_f32_16x16x32_f16(false, a, false, b,
                                                     (short)0, acc, false, false);
        if (t + 1 < nt) stB(t + 1);      // write other LDS buffer
        __syncthreads();
    }

    // ---- fused epilogue (exact tiles: no M/N guards) ----
    const int n = Nb + wv * 16 + l16;
#pragma unroll
    for (int r = 0; r < 8; ++r) {
        const int m = Mb + r + (hi << 3);   // D layout: vgpr r -> M=r (+8 hi lanes)
        float v = acc[r];
        if (bias) v += bias[m];
        if (act)  v = gelu_f(v);
        const long long off = (long long)m * N + n;
        if (R) v += R[off];
        Y[off] = v;
    }
}

// ---------------------------------------------------------------------------
// attn_raw[c,d] = sum_n hq[c,n]*k[d,n]  (M=N=48, K=HW), split-K WMMA + atomics
// grid: (9 tiles, NB*heads); 8 waves each own an interleaved K slice.
// HW is a multiple of 256 at every scale.
// ---------------------------------------------------------------------------
__global__ __launch_bounds__(256)
void k_attn_qk(const float* __restrict__ Q, long long qbs,
               const float* __restrict__ Kt, long long kbs,
               float* __restrict__ raw, int C, int heads, int HW)
{
    const int by = blockIdx.y;
    const int b = by / heads, h = by % heads;
    const int mt = blockIdx.x / 3, nt = blockIdx.x % 3;
    const int wv = threadIdx.x >> 5, lane = threadIdx.x & 31;
    const int l16 = lane & 15, hi = lane >> 4, kb = hi * 8;

    const float* qrow = Q  + (long long)b * qbs + ((long long)(h * 48 + mt * 16 + l16)) * HW;
    const float* krow = Kt + (long long)b * kbs + ((long long)(h * 48 + nt * 16 + l16)) * HW;

    v8f acc = {};
    for (int n = wv * 32; n < HW; n += 256) {
        v16h a, bf;
#pragma unroll
        for (int e = 0; e < 8; ++e) {
            a[e]      = (_Float16)qrow[n + kb + e];
            a[e + 8]  = (_Float16)qrow[n + 16 + kb + e];
            bf[e]     = (_Float16)krow[n + kb + e];
            bf[e + 8] = (_Float16)krow[n + 16 + kb + e];
        }
        acc = __builtin_amdgcn_wmma_f32_16x16x32_f16(false, a, false, bf,
                                                     (short)0, acc, false, false);
    }
    float* out = raw + (long long)by * 2304;
#pragma unroll
    for (int r = 0; r < 8; ++r) {
        const int mi = mt * 16 + r + hi * 8;
        const int ni = nt * 16 + l16;
        atomicAdd(&out[mi * 48 + ni], acc[r]);
    }
}

// attn = relu( raw / (||q_row|| * ||k_row||) * temp[h] )
__global__ __launch_bounds__(256)
void k_attn_fin(float* __restrict__ attn, const float* __restrict__ nq,
                const float* __restrict__ nk, const float* __restrict__ temp,
                int C, int heads)
{
    const int by = blockIdx.x;
    const int b = by / heads, h = by % heads;
    const float t = temp[h];
    const float* nqb = nq + (long long)b * C + h * 48;
    const float* nkb = nk + (long long)b * C + h * 48;
    float* a = attn + (long long)by * 2304;
    for (int i = threadIdx.x; i < 2304; i += 256) {
        const int r = i / 48, cc = i - r * 48;
        const float v = a[i] / (nqb[r] * nkb[cc]) * t;
        a[i] = v > 0.0f ? v : 0.0f;
    }
}

// row L2 norms over HW; out[b*C+c] = max(||row||, 1e-12)
__global__ __launch_bounds__(256)
void k_rownorm(const float* __restrict__ X, long long xbs,
               float* __restrict__ out, int HW, int C)
{
    const int c = blockIdx.x, b = blockIdx.y;
    const float* row = X + (long long)b * xbs + (long long)c * HW;
    float s = 0.0f;
    for (int i = threadIdx.x; i < HW; i += 256) { const float v = row[i]; s += v * v; }
    __shared__ float red[256];
    red[threadIdx.x] = s; __syncthreads();
    for (int st = 128; st > 0; st >>= 1) {
        if (threadIdx.x < st) red[threadIdx.x] += red[threadIdx.x + st];
        __syncthreads();
    }
    if (threadIdx.x == 0) {
        const float n = sqrtf(red[0]);
        out[b * C + c] = n > 1e-12f ? n : 1e-12f;
    }
}

// per-pixel LayerNorm over channel dim (NCHW, reduce over C)
__global__ __launch_bounds__(256)
void k_layernorm(const float* __restrict__ X, const float* __restrict__ g,
                 const float* __restrict__ bt, float* __restrict__ Y,
                 int C, int HW)
{
    const long long base = (long long)blockIdx.y * C * HW;
    const int n = blockIdx.x * 256 + threadIdx.x;
    if (n >= HW) return;
    const float* x = X + base + n;
    float mu = 0.0f;
    for (int c = 0; c < C; ++c) mu += x[(long long)c * HW];
    mu /= (float)C;
    float var = 0.0f;
    for (int c = 0; c < C; ++c) { const float d = x[(long long)c * HW] - mu; var += d * d; }
    var /= (float)C;
    const float inv = rsqrtf(var + 1e-5f);
    float* y = Y + base + n;
    for (int c = 0; c < C; ++c)
        y[(long long)c * HW] = (x[(long long)c * HW] - mu) * inv * g[c] + bt[c];
}

// depthwise 3x3, stride 1, pad = dil; optional bias, "- Sub", GELU
__global__ __launch_bounds__(256)
void k_dwconv3(const float* __restrict__ X, long long xbs,
               const float* __restrict__ Wm, const float* __restrict__ bias,
               const float* __restrict__ Sub, long long sbs,
               float* __restrict__ Y, long long ybs,
               int C, int H, int W, int dil, int act)
{
    const int c = blockIdx.y, b = blockIdx.z, HW = H * W;
    const int n = blockIdx.x * 256 + threadIdx.x;
    if (n >= HW) return;
    const int py = n / W, px = n - py * W;
    const float* in = X + (long long)b * xbs + (long long)c * HW;
    const float* w9 = Wm + c * 9;
    float acc = bias ? bias[c] : 0.0f;
#pragma unroll
    for (int ky = 0; ky < 3; ++ky)
#pragma unroll
        for (int kx = 0; kx < 3; ++kx) {
            const int yy = py + (ky - 1) * dil, xx = px + (kx - 1) * dil;
            if (yy >= 0 && yy < H && xx >= 0 && xx < W)
                acc += w9[ky * 3 + kx] * in[yy * W + xx];
        }
    if (Sub) acc -= Sub[(long long)b * sbs + (long long)c * HW + n];
    if (act) acc = gelu_f(acc);
    Y[(long long)b * ybs + (long long)c * HW + n] = acc;
}

__global__ __launch_bounds__(256)
void k_maxpool3(const float* __restrict__ X, long long xbs,
                float* __restrict__ Y, long long ybs, int C, int H, int W)
{
    const int c = blockIdx.y, b = blockIdx.z, HW = H * W;
    const int n = blockIdx.x * 256 + threadIdx.x;
    if (n >= HW) return;
    const int py = n / W, px = n - py * W;
    const float* in = X + (long long)b * xbs + (long long)c * HW;
    float m = -__builtin_inff();
#pragma unroll
    for (int dy = -1; dy <= 1; ++dy)
#pragma unroll
        for (int dx = -1; dx <= 1; ++dx) {
            const int yy = py + dy, xx = px + dx;
            if (yy >= 0 && yy < H && xx >= 0 && xx < W) {
                const float v = in[yy * W + xx];
                m = v > m ? v : m;
            }
        }
    Y[(long long)b * ybs + (long long)c * HW + n] = m;
}

__global__ __launch_bounds__(256)
void k_avgpool2(const float* __restrict__ X, float* __restrict__ Y, int C, int H, int W)
{
    const int Ho = H >> 1, Wo = W >> 1, HoWo = Ho * Wo;
    const int c = blockIdx.y, b = blockIdx.z;
    const int n = blockIdx.x * 256 + threadIdx.x;
    if (n >= HoWo) return;
    const int oy = n / Wo, ox = n - oy * Wo;
    const float* in = X + ((long long)b * C + c) * H * W;
    const float s = in[(2 * oy) * W + 2 * ox] + in[(2 * oy) * W + 2 * ox + 1] +
                    in[(2 * oy + 1) * W + 2 * ox] + in[(2 * oy + 1) * W + 2 * ox + 1];
    Y[((long long)b * C + c) * HoWo + n] = s * 0.25f;
}

// stem: 3x3 stride 2 pad 1, Cin=3, no bias
__global__ __launch_bounds__(256)
void k_embed(const float* __restrict__ X, const float* __restrict__ Wm,
             float* __restrict__ Y, int Ci, int Co, int Hi, int Wi)
{
    const int co = blockIdx.y, b = blockIdx.z;
    const int Ho = Hi >> 1, Wo = Wi >> 1;
    const int n = blockIdx.x * 256 + threadIdx.x;
    if (n >= Ho * Wo) return;
    const int oy = n / Wo, ox = n - oy * Wo;
    const int iy0 = oy * 2 - 1, ix0 = ox * 2 - 1;
    const float* xb = X + (long long)b * Ci * Hi * Wi;
    const float* wb = Wm + (long long)co * Ci * 9;
    float acc = 0.0f;
    for (int ci = 0; ci < Ci; ++ci)
#pragma unroll
        for (int ky = 0; ky < 3; ++ky)
#pragma unroll
            for (int kx = 0; kx < 3; ++kx) {
                const int yy = iy0 + ky, xx = ix0 + kx;
                if (yy >= 0 && yy < Hi && xx >= 0 && xx < Wi)
                    acc += wb[ci * 9 + ky * 3 + kx] *
                           xb[(long long)ci * Hi * Wi + yy * Wi + xx];
            }
    Y[((long long)b * Co + co) * Ho * Wo + n] = acc;
}

__global__ __launch_bounds__(256)
void k_gelu(const float* __restrict__ X, float* __restrict__ Y, long long n)
{
    const long long i = (long long)blockIdx.x * 256 + threadIdx.x;
    if (i < n) Y[i] = gelu_f(X[i]);
}

__global__ __launch_bounds__(256)
void k_zero(float* __restrict__ Y, long long n)
{
    const long long i = (long long)blockIdx.x * 256 + threadIdx.x;
    if (i < n) Y[i] = 0.0f;
}

// ===========================================================================
// Host orchestration
// ===========================================================================
struct FD  { const float *pw_w, *pw_b, *dw_w, *dw_b, *d1_w, *d1_b, *d2_w, *d2_b; };
struct Blk { const float *n1w, *n1b, *n2w, *n2b,
                         *qkv_w, *adw_w, *mp_w, *mp_b, *temp, *proj_w,
                         *fc1w, *fc1b, *fc2w, *fc2b; };

static inline void gemm(hipStream_t s,
                        const float* X, long long xb1, long long xb2,
                        const float* W, long long wb1, long long wb2,
                        const float* bias,
                        const float* R, long long rb1, long long rb2,
                        float* Y, long long yb1, long long yb2,
                        int M, int K, int N, int Z, int zdiv, int act)
{
    dim3 g(N / 128, M / 16, Z);   // exact: N%128==0, M%16==0 at every call site
    k_gemm16<<<g, 256, 0, s>>>(X, xb1, xb2, W, wb1, wb2, bias,
                               R, rb1, rb2, Y, yb1, yb2, M, K, N, zdiv, act);
}

static void freq(hipStream_t s, const float* xin, const FD& f, int C, int H, int W,
                 float* bufA, float* bufB, float* loC, float* loOut, float* hiOut)
{
    const int HW = H * W;
    const long long t = (long long)C * HW;
    // pw: c -> 2c, bias
    gemm(s, xin, t, 0, f.pw_w, 0, 0, f.pw_b, nullptr, 0, 0,
         bufA, 2 * t, 0, 2 * C, C, HW, NB, 1, 0);
    // dw 3x3 on 2c channels, bias
    dim3 g2c((HW + 255) / 256, 2 * C, NB);
    k_dwconv3<<<g2c, 256, 0, s>>>(bufA, 2 * t, f.dw_w, f.dw_b, nullptr, 0,
                                  bufB, 2 * t, 2 * C, H, W, 1, 0);
    dim3 gc((HW + 255) / 256, C, NB);
    // lo_pre = dwconv(x1, d2, pad2 dil2) + bias
    k_dwconv3<<<gc, 256, 0, s>>>(bufB, 2 * t, f.d2_w, f.d2_b, nullptr, 0,
                                 loC, t, C, H, W, 2, 0);
    // hi = gelu(dwconv(x2, d1) + bias - lo_pre)
    k_dwconv3<<<gc, 256, 0, s>>>(bufB + t, 2 * t, f.d1_w, f.d1_b, loC, t,
                                 hiOut, t, C, H, W, 1, 1);
    // lo = gelu(lo_pre)
    const long long n = (long long)NB * t;
    k_gelu<<<dim3((unsigned)((n + 255) / 256)), 256, 0, s>>>(loC, loOut, n);
}

static void mixer(hipStream_t s, float* X, const Blk& p, int C, int H, int W, int heads,
                  float* bufA, float* bufB, float* bufD, float* bufE,
                  float* nq, float* nk, float* attn)
{
    const int HW = H * W;
    const long long t = (long long)C * HW;
    dim3 gln((HW + 255) / 256, NB);
    // y = LN1(x)
    k_layernorm<<<gln, 256, 0, s>>>(X, p.n1w, p.n1b, bufD, C, HW);
    // qkv = 1x1 (no bias)
    gemm(s, bufD, t, 0, p.qkv_w, 0, 0, nullptr, nullptr, 0, 0,
         bufA, 3 * t, 0, 3 * C, C, HW, NB, 1, 0);
    // depthwise 3x3 on 3c (no bias)
    dim3 g3c((HW + 255) / 256, 3 * C, NB);
    k_dwconv3<<<g3c, 256, 0, s>>>(bufA, 3 * t, p.adw_w, nullptr, nullptr, 0,
                                  bufB, 3 * t, 3 * C, H, W, 1, 0);
    // q = bufB+0, k = bufB+t, v = bufB+2t (per-batch stride 3t)
    dim3 gc((HW + 255) / 256, C, NB);
    k_maxpool3<<<gc, 256, 0, s>>>(bufB, 3 * t, bufD, t, C, H, W);
    // hq = gelu(1x1(mq) + bias)
    gemm(s, bufD, t, 0, p.mp_w, 0, 0, p.mp_b, nullptr, 0, 0,
         bufE, t, 0, C, C, HW, NB, 1, 1);
    // row norms (l2norm folded into attention epilogue)
    k_rownorm<<<dim3(C, NB), 256, 0, s>>>(bufE, t, nq, HW, C);
    k_rownorm<<<dim3(C, NB), 256, 0, s>>>(bufB + t, 3 * t, nk, HW, C);
    // attn_raw = hq @ k^T  (split-K WMMA, atomics)
    const long long araw = (long long)NB * heads * 2304;
    k_zero<<<dim3((unsigned)((araw + 255) / 256)), 256, 0, s>>>(attn, araw);
    k_attn_qk<<<dim3(9, NB * heads), 256, 0, s>>>(bufE, t, bufB + t, 3 * t,
                                                  attn, C, heads, HW);
    k_attn_fin<<<dim3(NB * heads), 256, 0, s>>>(attn, nq, nk, p.temp, C, heads);
    // out2 = attn @ v   (per-(b,h) GEMM via zdiv=heads)
    gemm(s, bufB + 2 * t, 3 * t, 48LL * HW,
         attn, (long long)heads * 2304, 2304, nullptr,
         nullptr, 0, 0,
         bufD, t, 48LL * HW,
         48, 48, HW, NB * heads, heads, 0);
    // x = proj(out2) + x   (residual fused, in place)
    gemm(s, bufD, t, 0, p.proj_w, 0, 0, nullptr, X, t, 0, X, t, 0,
         C, C, HW, NB, 1, 0);
    // FFN
    k_layernorm<<<gln, 256, 0, s>>>(X, p.n2w, p.n2b, bufD, C, HW);
    gemm(s, bufD, t, 0, p.fc1w, 0, 0, p.fc1b, nullptr, 0, 0,
         bufA, 2 * t, 0, 2 * C, C, HW, NB, 1, 1);
    gemm(s, bufA, 2 * t, 0, p.fc2w, 0, 0, p.fc2b, X, t, 0, X, t, 0,
         C, 2 * C, HW, NB, 1, 0);
}

extern "C" void kernel_launch(void* const* d_in, const int* in_sizes, int n_in,
                              void* d_out, int out_size, void* d_ws, size_t ws_size,
                              hipStream_t stream)
{
    (void)in_sizes; (void)n_in; (void)out_size; (void)ws_size;
    // ---- unpack inputs (setup_inputs() insertion order, recursively) ----
    int ip = 0;
    auto nx = [&]() -> const float* { return (const float*)d_in[ip++]; };
    const float* x       = nx();
    const float* embed_w = nx();
    FD fd[3];
    for (int i = 0; i < 3; ++i) {
        fd[i].pw_w = nx(); fd[i].pw_b = nx(); fd[i].dw_w = nx(); fd[i].dw_b = nx();
        fd[i].d1_w = nx(); fd[i].d1_b = nx(); fd[i].d2_w = nx(); fd[i].d2_b = nx();
    }
    const float* inc1_w = nx(); const float* inc1_b = nx();
    const float* inc2_w = nx(); const float* inc2_b = nx();
    Blk hf0[2], hf1[4], hf2[4];
    auto rdblk = [&](Blk& b) {
        b.n1w = nx(); b.n1b = nx(); b.n2w = nx(); b.n2b = nx();
        b.qkv_w = nx(); b.adw_w = nx(); b.mp_w = nx(); b.mp_b = nx();
        b.temp = nx(); b.proj_w = nx();
        b.fc1w = nx(); b.fc1b = nx(); b.fc2w = nx(); b.fc2b = nx();
    };
    for (int i = 0; i < 2; ++i) rdblk(hf0[i]);
    for (int i = 0; i < 4; ++i) rdblk(hf1[i]);
    for (int i = 0; i < 4; ++i) rdblk(hf2[i]);

    // ---- geometry ----
    const int C0 = 48,  H0 = 192, W0 = 192, HW0 = H0 * W0;
    const int C1 = 96,  H1 = 96,  W1 = 96,  HW1 = H1 * W1;
    const int C2 = 192, H2 = 48,  W2 = 48,  HW2 = H2 * W2;
    const long long T0 = (long long)NB * C0 * HW0;  // 7,077,888
    const long long T1 = (long long)NB * C1 * HW1;  // 3,538,944
    const long long T2 = (long long)NB * C2 * HW2;  // 1,769,472

    // ---- workspace layout (floats); total ~= 10*T0 + 64K ~= 284 MB ----
    float* WS   = (float*)d_ws;
    float* bufA = WS;             // 3*T0
    float* bufB = WS + 3 * T0;    // 3*T0
    float* bufC = WS + 6 * T0;    // 2*T0
    float* bufD = WS + 8 * T0;    // T0
    float* bufE = WS + 9 * T0;    // T0
    float* nq   = WS + 10 * T0;
    float* nk   = nq + 4096;
    float* attn = nk + 4096;      // <= 16*2304 floats

    // ---- output slices ----
    float* out  = (float*)d_out;
    float* hi0  = out;
    float* hi1  = out + T0;
    float* hi2  = out + T0 + T1;
    float* lo2  = out + T0 + T1 + T2;

    // ---- stem ----
    k_embed<<<dim3((HW0 + 255) / 256, C0, NB), 256, 0, stream>>>(
        x, embed_w, bufD, 3, C0, 384, 384);

    // ---- freqdec 0 ----
    freq(stream, bufD, fd[0], C0, H0, W0, bufA, bufB, bufC, bufC, hi0);
    // x1 = inc1(avgpool2(lo0))
    k_avgpool2<<<dim3((HW1 + 255) / 256, C0, NB), 256, 0, stream>>>(bufC, bufE, C0, H0, W0);
    gemm(stream, bufE, (long long)C0 * HW1, 0, inc1_w, 0, 0, inc1_b,
         nullptr, 0, 0, bufD, (long long)C1 * HW1, 0, C1, C0, HW1, NB, 1, 0);

    // ---- freqdec 1 ----
    freq(stream, bufD, fd[1], C1, H1, W1, bufA, bufB, bufC, bufC, hi1);
    k_avgpool2<<<dim3((HW2 + 255) / 256, C1, NB), 256, 0, stream>>>(bufC, bufE, C1, H1, W1);
    gemm(stream, bufE, (long long)C1 * HW2, 0, inc2_w, 0, 0, inc2_b,
         nullptr, 0, 0, bufD, (long long)C2 * HW2, 0, C2, C1, HW2, NB, 1, 0);

    // ---- freqdec 2 (lo2 goes straight to d_out) ----
    freq(stream, bufD, fd[2], C2, H2, W2, bufA, bufB, bufC, lo2, hi2);

    // ---- highpass mixers (in place in d_out) ----
    for (int i = 0; i < 2; ++i)
        mixer(stream, hi0, hf0[i], C0, H0, W0, 1, bufA, bufB, bufD, bufE, nq, nk, attn);
    for (int i = 0; i < 4; ++i)
        mixer(stream, hi1, hf1[i], C1, H1, W1, 2, bufA, bufB, bufD, bufE, nq, nk, attn);
    for (int i = 0; i < 4; ++i)
        mixer(stream, hi2, hf2[i], C2, H2, W2, 4, bufA, bufB, bufD, bufE, nq, nk, attn);
}
